// Block_36060545417393
// MI455X (gfx1250) — matrix operand order repaired and checked by reading, not compiled, not run on hardware
//
#include <hip/hip_runtime.h>

#define DEV_INLINE __device__ __forceinline__

typedef __attribute__((ext_vector_type(16))) __bf16 v16bf;
typedef __attribute__((ext_vector_type(8)))  __bf16 v8bf;
typedef __attribute__((ext_vector_type(8)))  float  v8f;

// Problem constants (reference: B=2, T=2048, D=768, H=12, HD=64)
constexpr int BB = 2;
constexpr int TT = 2048;
constexpr int DD = 768;
constexpr int HH = 12;
constexpr int HD = 64;
constexpr int MR = BB * TT;      // 4096 rows

enum { EPI_HEAD = 0, EPI_HEADT = 1, EPI_RELU = 2, EPI_RES = 3 };

DEV_INLINE v16bf cat8(v8bf lo, v8bf hi) {
  v16bf r;
#pragma unroll
  for (int i = 0; i < 8; ++i) { r[i] = lo[i]; r[i + 8] = hi[i]; }
  return r;
}

DEV_INLINE v8f wmma_bf16(v16bf a, v16bf b, v8f c) {
  return __builtin_amdgcn_wmma_f32_16x16x32_bf16(
      /*neg_a=*/false, a, /*neg_b=*/false, b,
      /*c_mod=*/(short)0, c, /*reuse_a=*/false, /*reuse_b=*/false);
}

// ---------------------------------------------------------------------------
// Weight convert + transpose: W[K,N] fp32 -> Wt[N,K] bf16
// ---------------------------------------------------------------------------
__global__ void tconv_kernel(const float* __restrict__ W, __bf16* __restrict__ Wt,
                             int K, int N) {
  int i = blockIdx.x * 256 + threadIdx.x;
  if (i >= K * N) return;
  int k = i / N, n = i % N;
  Wt[(size_t)n * K + k] = (__bf16)W[i];
}

// ---------------------------------------------------------------------------
// LayerNorm: x[row, 768] fp32 -> out bf16   (one 256-thread block per row)
// ---------------------------------------------------------------------------
__global__ void ln_bf16_kernel(const float* __restrict__ x, const float* __restrict__ g,
                               const float* __restrict__ b, __bf16* __restrict__ out) {
  __shared__ float s1[256], s2[256];
  int row = blockIdx.x, tid = threadIdx.x;
  const float* xr = x + (size_t)row * DD;
  float v0 = xr[tid], v1 = xr[tid + 256], v2 = xr[tid + 512];
  s1[tid] = v0 + v1 + v2;
  s2[tid] = v0 * v0 + v1 * v1 + v2 * v2;
  __syncthreads();
  for (int off = 128; off > 0; off >>= 1) {
    if (tid < off) { s1[tid] += s1[tid + off]; s2[tid] += s2[tid + off]; }
    __syncthreads();
  }
  float mean = s1[0] * (1.0f / DD);
  float var  = s2[0] * (1.0f / DD) - mean * mean;
  float rstd = rsqrtf(var + 1e-5f);
  __bf16* o = out + (size_t)row * DD;
  o[tid]       = (__bf16)((v0 - mean) * rstd * g[tid]       + b[tid]);
  o[tid + 256] = (__bf16)((v1 - mean) * rstd * g[tid + 256] + b[tid + 256]);
  o[tid + 512] = (__bf16)((v2 - mean) * rstd * g[tid + 512] + b[tid + 512]);
}

// ---------------------------------------------------------------------------
// bf16 GEMM: C[M,N] = A[M,K] @ W[K,N], W supplied transposed (Bt[N,K]).
// One wave -> 32x64 output tile: 2 A-fragments x 4 B-fragments = 8 WMMAs
// per K-step of 32 (24 B of L2 traffic per WMMA, 8 independent chains).
// Epilogue modes: scatter to head layout (Q/K), transposed head layout (V),
// ReLU->bf16 (MLP1), fp32 residual add (Wo / MLP2).
// ---------------------------------------------------------------------------
__global__ void gemm_bf16_kernel(const __bf16* __restrict__ A, const __bf16* __restrict__ Bt,
                                 int M, int N, int K, int mode, float scale,
                                 const float* __restrict__ res, float* __restrict__ outF,
                                 __bf16* __restrict__ outB) {
  const int wave   = threadIdx.x >> 5;
  const int lane   = threadIdx.x & 31;
  const int laneHi = lane >> 4;
  const int l16    = lane & 15;

  const int tile   = blockIdx.x * 8 + wave;
  const int mTiles = M >> 5;                 // 32-row tiles
  const int m0 = (tile % mTiles) << 5;
  const int n0 = (tile / mTiles) << 6;

  v8f acc[2][4];
#pragma unroll
  for (int a = 0; a < 2; ++a)
#pragma unroll
    for (int nt = 0; nt < 4; ++nt)
#pragma unroll
      for (int r = 0; r < 8; ++r) acc[a][nt][r] = 0.0f;

  const __bf16* arow0 = A + (size_t)(m0 + l16) * K;
  const __bf16* arow1 = A + (size_t)(m0 + 16 + l16) * K;
  for (int k0 = 0; k0 < K; k0 += 32) {
    v16bf a0 = cat8(*reinterpret_cast<const v8bf*>(arow0 + k0 + 16 * laneHi),
                    *reinterpret_cast<const v8bf*>(arow0 + k0 + 8 + 16 * laneHi));
    v16bf a1 = cat8(*reinterpret_cast<const v8bf*>(arow1 + k0 + 16 * laneHi),
                    *reinterpret_cast<const v8bf*>(arow1 + k0 + 8 + 16 * laneHi));
#pragma unroll
    for (int nt = 0; nt < 4; ++nt) {
      const __bf16* bcol = Bt + (size_t)(n0 + nt * 16 + l16) * K + k0 + 16 * laneHi;
      v16bf b = *reinterpret_cast<const v16bf*>(bcol);
      acc[0][nt] = wmma_bf16(a0, b, acc[0][nt]);
      acc[1][nt] = wmma_bf16(a1, b, acc[1][nt]);
    }
  }

#pragma unroll
  for (int a = 0; a < 2; ++a) {
#pragma unroll
    for (int nt = 0; nt < 4; ++nt) {
#pragma unroll
      for (int r = 0; r < 8; ++r) {
        float v = acc[a][nt][r];
        int m = m0 + a * 16 + r + 8 * laneHi;  // C layout: VGPR r -> M = r + 8*(lane>=16)
        int n = n0 + nt * 16 + l16;            // N = lane%16
        if (mode == EPI_RES) {
          size_t idx = (size_t)m * N + n;
          outF[idx] = v + res[idx];
        } else if (mode == EPI_RELU) {
          outB[(size_t)m * N + n] = (__bf16)fmaxf(v, 0.0f);
        } else {
          int t = m & (TT - 1), bb = m >> 11;   // T = 2048
          int h = n >> 6, hd = n & (HD - 1);
          size_t idx;
          if (mode == EPI_HEAD) idx = (((size_t)(bb * HH + h)) * TT + t) * HD + hd;
          else                  idx = (((size_t)(bb * HH + h)) * HD + hd) * TT + t;
          outB[idx] = (__bf16)(v * scale);
        }
      }
    }
  }
}

// ---------------------------------------------------------------------------
// Flash attention (causal). q,k in [B,H,T,HD] bf16 (q pre-scaled by HD^-0.5),
// v stored transposed [B,H,HD,T] bf16. Computes S^T = K*Q^T so softmax is a
// per-lane column reduce, and P^T feeds O^T += V^T * P^T directly as the WMMA
// B operand (register permute + cross-half shuffles, no LDS transpose).
// One wave per 16 query rows; block = 8 waves = 128 rows.
// ---------------------------------------------------------------------------
__global__ void attn_kernel(const __bf16* __restrict__ qh, const __bf16* __restrict__ kh,
                            const __bf16* __restrict__ vt, __bf16* __restrict__ obf) {
  const int wave   = threadIdx.x >> 5;
  const int lane   = threadIdx.x & 31;
  const int laneHi = lane >> 4;
  const int l16    = lane & 15;

  const int qBlk = blockIdx.x % (TT / 128);
  const int bh   = blockIdx.x / (TT / 128);
  const int q0   = qBlk * 128 + wave * 16;
  const int qi   = q0 + l16;            // this lane's query column

  const __bf16* qp = qh + (size_t)bh * TT * HD;
  const __bf16* kp = kh + (size_t)bh * TT * HD;
  const __bf16* vp = vt + (size_t)bh * HD * TT;

  // Q^T B-fragments (fixed for the whole key loop): n = query col, k = hd
  const __bf16* qrow = qp + (size_t)qi * HD;
  v16bf bq0 = cat8(*reinterpret_cast<const v8bf*>(qrow + 8 * laneHi),
                   *reinterpret_cast<const v8bf*>(qrow + 16 + 8 * laneHi));        // hd 0..31
  v16bf bq1 = cat8(*reinterpret_cast<const v8bf*>(qrow + 32 + 8 * laneHi),
                   *reinterpret_cast<const v8bf*>(qrow + 48 + 8 * laneHi));   // hd 32..63

  v8f O[4];
#pragma unroll
  for (int t = 0; t < 4; ++t)
#pragma unroll
    for (int r = 0; r < 8; ++r) O[t][r] = 0.0f;

  float m_run = -1e30f, l_run = 0.0f;
  const int nCh = (q0 + 15) / 32 + 1;   // key chunks of 32 covering keys <= q0+15

  for (int ch = 0; ch < nCh; ++ch) {
    const int c0 = ch * 32;
    // ----- scores S^T (two 16-key x 16-query tiles), K-dim = hd (64) -----
    v8f S0, S1;
#pragma unroll
    for (int r = 0; r < 8; ++r) { S0[r] = 0.0f; S1[r] = 0.0f; }
    {
      const __bf16* kb0 = kp + (size_t)(c0 + l16) * HD;
      const __bf16* kb1 = kp + (size_t)(c0 + 16 + l16) * HD;
      v16bf a00 = cat8(*reinterpret_cast<const v8bf*>(kb0 + 8 * laneHi),
                       *reinterpret_cast<const v8bf*>(kb0 + 16 + 8 * laneHi));
      v16bf a01 = cat8(*reinterpret_cast<const v8bf*>(kb0 + 32 + 8 * laneHi),
                       *reinterpret_cast<const v8bf*>(kb0 + 48 + 8 * laneHi));
      v16bf a10 = cat8(*reinterpret_cast<const v8bf*>(kb1 + 8 * laneHi),
                       *reinterpret_cast<const v8bf*>(kb1 + 16 + 8 * laneHi));
      v16bf a11 = cat8(*reinterpret_cast<const v8bf*>(kb1 + 32 + 8 * laneHi),
                       *reinterpret_cast<const v8bf*>(kb1 + 48 + 8 * laneHi));
      S0 = wmma_bf16(a00, bq0, S0); S0 = wmma_bf16(a01, bq1, S0);
      S1 = wmma_bf16(a10, bq0, S1); S1 = wmma_bf16(a11, bq1, S1);
    }
    // ----- causal mask (select, keeps EXEC all-ones) -----
    if (c0 + 31 > q0) {
#pragma unroll
      for (int r = 0; r < 8; ++r) {
        int key0 = c0 + r + 8 * laneHi;
        S0[r] = (key0 > qi) ? -1e30f : S0[r];
        S1[r] = (key0 + 16 > qi) ? -1e30f : S1[r];
      }
    }
    // ----- online softmax (per-lane column stats, one cross-half shuffle) -----
    float cm = S0[0];
#pragma unroll
    for (int r = 1; r < 8; ++r) cm = fmaxf(cm, S0[r]);
#pragma unroll
    for (int r = 0; r < 8; ++r) cm = fmaxf(cm, S1[r]);
    cm = fmaxf(cm, __shfl_xor(cm, 16));
    float mnew = fmaxf(m_run, cm);
    float corr = __expf(m_run - mnew);
    l_run *= corr;
#pragma unroll
    for (int t = 0; t < 4; ++t)
#pragma unroll
      for (int r = 0; r < 8; ++r) O[t][r] *= corr;
    float ps = 0.0f;
#pragma unroll
    for (int r = 0; r < 8; ++r) { S0[r] = __expf(S0[r] - mnew); ps += S0[r]; }
#pragma unroll
    for (int r = 0; r < 8; ++r) { S1[r] = __expf(S1[r] - mnew); ps += S1[r]; }
    ps += __shfl_xor(ps, 16);
    l_run += ps;
    m_run = mnew;
    // ----- assemble P^T as WMMA B operand: k(key) = j + 16*laneHi -----
    v16bf bp;
#pragma unroll
    for (int j = 0; j < 8; ++j) {
      float a = S0[j], b = S1[j];
      float ax = __shfl_xor(a, 16), bx = __shfl_xor(b, 16);
      bp[j]     = (__bf16)(laneHi ? bx : a);
      bp[j + 8] = (__bf16)(laneHi ? b : ax);
    }
    // ----- O^T += V^T @ P^T -----
#pragma unroll
    for (int t = 0; t < 4; ++t) {
      const __bf16* vrow = vp + (size_t)(t * 16 + l16) * TT + c0;
      v16bf av = cat8(*reinterpret_cast<const v8bf*>(vrow + 16 * laneHi),
                      *reinterpret_cast<const v8bf*>(vrow + 8 + 16 * laneHi));
      O[t] = wmma_bf16(av, bp, O[t]);
    }
  }

  // ----- normalize and write o[b, t, h*64 + hd] bf16 -----
  float inv = 1.0f / l_run;
  const int b = bh / HH, h = bh % HH;
  const size_t rowBase = ((size_t)b * TT + qi) * DD + h * HD;
#pragma unroll
  for (int t = 0; t < 4; ++t) {
    v8bf ov;
#pragma unroll
    for (int r = 0; r < 8; ++r) ov[r] = (__bf16)(O[t][r] * inv);
    *reinterpret_cast<v8bf*>(obf + rowBase + t * 16 + 8 * laneHi) = ov;
  }
}

// ---------------------------------------------------------------------------
// Host-side orchestration
// ---------------------------------------------------------------------------
extern "C" void kernel_launch(void* const* d_in, const int* in_sizes, int n_in,
                              void* d_out, int out_size, void* d_ws, size_t ws_size,
                              hipStream_t stream) {
  const float* x  = (const float*)d_in[0];
  const float* Wq = (const float*)d_in[1];
  const float* Wk = (const float*)d_in[2];
  const float* Wv = (const float*)d_in[3];
  const float* Wo = (const float*)d_in[4];
  const float* W1 = (const float*)d_in[5];
  const float* W2 = (const float*)d_in[6];
  const float* g1 = (const float*)d_in[7];
  const float* b1 = (const float*)d_in[8];
  const float* g2 = (const float*)d_in[9];
  const float* b2 = (const float*)d_in[10];
  float* out = (float*)d_out;

  char* ws = (char*)d_ws;
  size_t off = 0;
  auto alloc = [&](size_t bytes) { char* p = ws + off; off += (bytes + 255) & ~size_t(255); return p; };
  const size_t actB = (size_t)MR * DD * 2;          // 4096x768 bf16
  __bf16* h1  = (__bf16*)alloc(actB);
  __bf16* wqt = (__bf16*)alloc((size_t)DD * DD * 2);
  __bf16* wkt = (__bf16*)alloc((size_t)DD * DD * 2);
  __bf16* wvt = (__bf16*)alloc((size_t)DD * DD * 2);
  __bf16* wot = (__bf16*)alloc((size_t)DD * DD * 2);
  __bf16* w1t = (__bf16*)alloc((size_t)DD * 4 * DD * 2);
  __bf16* w2t = (__bf16*)alloc((size_t)4 * DD * DD * 2);
  __bf16* qhB = (__bf16*)alloc(actB);
  __bf16* khB = (__bf16*)alloc(actB);
  __bf16* vtB = (__bf16*)alloc(actB);
  __bf16* oB  = (__bf16*)alloc(actB);
  float*  x1  = (float*)alloc((size_t)MR * DD * 4);
  __bf16* h2  = (__bf16*)alloc(actB);
  __bf16* a1  = (__bf16*)alloc((size_t)MR * 4 * DD * 2);

  const dim3 blk(256);
  // weights -> bf16 transposed
  int nw = DD * DD;
  tconv_kernel<<<(nw + 255) / 256, blk, 0, stream>>>(Wq, wqt, DD, DD);
  tconv_kernel<<<(nw + 255) / 256, blk, 0, stream>>>(Wk, wkt, DD, DD);
  tconv_kernel<<<(nw + 255) / 256, blk, 0, stream>>>(Wv, wvt, DD, DD);
  tconv_kernel<<<(nw + 255) / 256, blk, 0, stream>>>(Wo, wot, DD, DD);
  int nw2 = DD * 4 * DD;
  tconv_kernel<<<(nw2 + 255) / 256, blk, 0, stream>>>(W1, w1t, DD, 4 * DD);
  tconv_kernel<<<(nw2 + 255) / 256, blk, 0, stream>>>(W2, w2t, 4 * DD, DD);

  // LN1
  ln_bf16_kernel<<<MR, blk, 0, stream>>>(x, g1, b1, h1);

  // QKV projections: grid = (M/32 * N/64) / 8 waves per block
  const int gQKV = (MR / 32) * (DD / 64) / 8;          // 192
  gemm_bf16_kernel<<<gQKV, blk, 0, stream>>>(h1, wqt, MR, DD, DD, EPI_HEAD, 0.125f,
                                             nullptr, nullptr, qhB);
  gemm_bf16_kernel<<<gQKV, blk, 0, stream>>>(h1, wkt, MR, DD, DD, EPI_HEAD, 1.0f,
                                             nullptr, nullptr, khB);
  gemm_bf16_kernel<<<gQKV, blk, 0, stream>>>(h1, wvt, MR, DD, DD, EPI_HEADT, 1.0f,
                                             nullptr, nullptr, vtB);

  // causal flash attention: B*H * (T/128) blocks
  attn_kernel<<<BB * HH * (TT / 128), blk, 0, stream>>>(qhB, khB, vtB, oB);

  // Wo projection + residual -> x1 (fp32)
  gemm_bf16_kernel<<<gQKV, blk, 0, stream>>>(oB, wot, MR, DD, DD, EPI_RES, 1.0f,
                                             x, x1, nullptr);
  // LN2
  ln_bf16_kernel<<<MR, blk, 0, stream>>>(x1, g2, b2, h2);

  // MLP1: 4096x3072, ReLU -> bf16
  const int gM1 = (MR / 32) * (4 * DD / 64) / 8;       // 768
  gemm_bf16_kernel<<<gM1, blk, 0, stream>>>(h2, w1t, MR, 4 * DD, DD, EPI_RELU, 1.0f,
                                            nullptr, nullptr, a1);
  // MLP2: K=3072, residual add -> d_out
  gemm_bf16_kernel<<<gQKV, blk, 0, stream>>>(a1, w2t, MR, DD, 4 * DD, EPI_RES, 1.0f,
                                             x1, out, nullptr);
  (void)in_sizes; (void)n_in; (void)out_size; (void)ws_size;
}
